// LocalDecoder_32006096290031
// MI455X (gfx1250) — matrix operand order repaired and checked by reading, not compiled, not run on hardware
//
#include <hip/hip_runtime.h>

// ---------------------------------------------------------------------------
// Problem constants
// ---------------------------------------------------------------------------
#define XD      130          // X_DIM
#define COND    44           // COND_DIM
#define IN0     174          // XD + COND
#define IN0P    192          // padded to multiple of 32 for K
#define H       1024
#define G4      4096         // 4*H
#define NSTEP   16
#define BP      4096         // effective batch (64*64)
#define TB      32           // batch rows per block (two WMMA M tiles)
#define XDP     144          // Wf N padded to multiple of 16
#define NTHR    512          // 16 waves per block

// ---------------------------------------------------------------------------
// WMMA types / helpers (gfx1250, wave32)
// ---------------------------------------------------------------------------
typedef __attribute__((ext_vector_type(16))) __bf16        v16bf;
typedef __attribute__((ext_vector_type(8)))  float         v8f;
typedef __attribute__((ext_vector_type(4)))  unsigned int  v4u;

union FragBF { v16bf bf; v4u u4[2]; };

__device__ __forceinline__ unsigned short bf16b(float f) {
  unsigned int u = __builtin_bit_cast(unsigned int, f);
  return (unsigned short)((u + 0x7FFFu + ((u >> 16) & 1u)) >> 16); // RNE
}

__device__ __forceinline__ float sigm(float x) {
  return 1.0f / (1.0f + __expf(-x));
}

__device__ __forceinline__ v8f wmma_bf16(v16bf a, v16bf b, v8f c) {
  // D = A(16x32) * B(32x16) + C, f32 accumulate
  return __builtin_amdgcn_wmma_f32_16x16x32_bf16(
      /*neg_a=*/false, a, /*neg_b=*/false, b,
      /*c_mod=*/(short)0, c, /*reuse_a=*/false, /*reuse_b=*/false);
}

// A fragment (16x32 bf16, M x K) from LDS, row-major with 'stride' elements.
// Lane l: M = l&15 ; K = (l>>4)*8 .. +7  and  (l>>4)*8+16 .. +23  (ISA 7.12.2)
__device__ __forceinline__ v16bf ld_a(const unsigned short* A, int stride,
                                      int lane, int k0) {
  FragBF f;
  const unsigned short* p = A + (lane & 15) * stride + ((lane >> 4) << 3) + k0;
  f.u4[0] = *(const v4u*)(p);
  f.u4[1] = *(const v4u*)(p + 16);
  return f.bf;
}

// B fragment (32x16 bf16, K x N) with B[k][n] = W[n][k]; Wrow0 = first weight
// row of the N-tile (row-major, 'stride' = padded K).
// Lane l: N = l&15 ; K = (l>>4)*8 .. +7  and  (l>>4)*8+16 .. +23 (matches ld_a).
__device__ __forceinline__ v16bf ld_b(const unsigned short* Wrow0, int stride,
                                      int lane, int k0) {
  FragBF f;
  const unsigned short* p =
      Wrow0 + (long)(lane & 15) * stride + ((lane >> 4) << 3) + k0;
  f.u4[0] = *(const v4u*)(p);
  f.u4[1] = *(const v4u*)(p + 16);
  return f.bf;
}

// K-sweep for one N-tile: 4 gate columns x 2 M-tiles (8 accumulators).
// Each B fragment (1KB of L2 traffic) feeds TWO WMMAs -> 32 FLOP/byte.
__device__ __forceinline__ void gemm8(v8f acc[8],
                                      const unsigned short* A, int astride,
                                      int kTiles, const unsigned short* W,
                                      int wstride, int col0, int lane) {
  const unsigned short* w[4];
  w[0] = W + (long)(col0)       * wstride;
  w[1] = W + (long)(H   + col0) * wstride;
  w[2] = W + (long)(2*H + col0) * wstride;
  w[3] = W + (long)(3*H + col0) * wstride;
  for (int kt = 0; kt < kTiles; ++kt) {
    const int k0 = kt * 32;
    v16bf a0 = ld_a(A,               astride, lane, k0);  // rows 0-15
    v16bf a1 = ld_a(A + 16 * astride, astride, lane, k0); // rows 16-31
#pragma unroll
    for (int q = 0; q < 4; ++q) {
      v16bf b = ld_b(w[q], wstride, lane, k0);
      acc[q]     = wmma_bf16(a0, b, acc[q]);
      acc[4 + q] = wmma_bf16(a1, b, acc[4 + q]);
    }
  }
}

// LSTM cell elementwise update for one 16x16 M-tile. C/D layout (ISA 7.12.2):
// VGPR r -> row rtop + r, col = lane&15. Cell state lives in global ws (f32,
// L2-resident; ~1 GB total traffic vs the ~50 GB of weight reads it frees
// from LDS capacity). t==0 uses literal zero (never reads poisoned ws).
__device__ __forceinline__ void cell_update(v8f g_i, v8f g_f, v8f g_g, v8f g_o,
                                            const float* __restrict__ bias,
                                            int ncol, int rtop, bool first,
                                            float* __restrict__ cg,
                                            long growbase,
                                            unsigned short* hwr) {
  const float bi = bias[ncol];
  const float bff = bias[H + ncol];
  const float bg = bias[2*H + ncol];
  const float bo = bias[3*H + ncol];
#pragma unroll
  for (int r = 0; r < 8; ++r) {
    const int rr = rtop + r;                      // 0..31 tile-local row
    const long gi = growbase + (long)rr * H + ncol;
    const float cv = first ? 0.0f : cg[gi];
    const float iv = sigm(g_i[r] + bi);
    const float fv = sigm(g_f[r] + bff);
    const float gv = tanhf(g_g[r] + bg);
    const float ov = sigm(g_o[r] + bo);
    const float c2 = fv * cv + iv * gv;
    cg[gi] = c2;
    hwr[rr * H + ncol] = bf16b(ov * tanhf(c2));
  }
}

// ---------------------------------------------------------------------------
// Prep kernels: f32 -> bf16 with K (and N) zero-padding; bias pre-sum.
// ---------------------------------------------------------------------------
__global__ void cvt_bf16_pad(const float* __restrict__ src,
                             unsigned short* __restrict__ dst,
                             int nsrc, int npad, int k, int kpad) {
  long i = (long)blockIdx.x * blockDim.x + threadIdx.x;
  long total = (long)npad * kpad;
  if (i >= total) return;
  int n = (int)(i / kpad);
  int kk = (int)(i % kpad);
  float v = (n < nsrc && kk < k) ? src[(long)n * k + kk] : 0.0f;
  dst[i] = bf16b(v);
}

__global__ void add_vec(const float* __restrict__ a,
                        const float* __restrict__ b,
                        float* __restrict__ o, int n) {
  int i = blockIdx.x * blockDim.x + threadIdx.x;
  if (i < n) o[i] = a[i] + b[i];
}

// ---------------------------------------------------------------------------
// Persistent decoder kernel: 128 blocks x 32 batch rows; 16 waves/block each
// owning a 64-wide hidden slice; all 16 time steps inside the kernel.
// h-state uses a 3-buffer LDS rotation (h0_old dies once layer 1 starts, so
// h1_new reuses its buffer): 3x64KB instead of 4x64KB double buffering.
// ---------------------------------------------------------------------------
__global__ __launch_bounds__(NTHR)
void lstm_dec(const float* __restrict__ z, const float* __restrict__ x,
              const unsigned short* __restrict__ w_ih0,
              const unsigned short* __restrict__ w_hh0,
              const unsigned short* __restrict__ w_ih1,
              const unsigned short* __restrict__ w_hh1,
              const unsigned short* __restrict__ wf,
              const float* __restrict__ bias0,
              const float* __restrict__ bias1,
              const float* __restrict__ bfv,
              float* __restrict__ c0g, float* __restrict__ c1g,
              float* __restrict__ out) {
  extern __shared__ __align__(16) char smem[];
  unsigned short* inp  = (unsigned short*)(smem);              // 32*192 bf16
  unsigned short* hbuf = (unsigned short*)(smem + 12288);      // 3 x 32*1024
  float* lgt  = (float*)(smem + 12288 + 196608);               // 32*144 f32
  float* rmax = (float*)(smem + 12288 + 196608 + 18432);       // 512 f32
  int*   ridx = (int*)  (smem + 12288 + 196608 + 18432 + 2048);// 512 i32
  float* rsum = (float*)(smem + 12288 + 196608 + 18432 + 4096);// 512 f32

  const int tid   = threadIdx.x;
  const int lane  = tid & 31;
  const int wave  = tid >> 5;          // 0..15
  const int rbase = blockIdx.x * TB;
  const int rowr  = tid >> 4;          // 0..31: reduction row mapping
  const int sub   = tid & 15;
  const int rtop  = (lane >> 4) << 3;  // C-tile top row for this lane half
  const long growbase = (long)rbase * H;

  // ---- init: h0=h1=z (buffers 0,1), inp = [one-hot(1), y_0, zero-pad] ----
  for (int i = tid; i < TB * H; i += NTHR) {
    unsigned short v = bf16b(z[(long)(rbase + (i >> 10)) * H + (i & (H - 1))]);
    hbuf[i] = v;
    hbuf[TB * H + i] = v;
  }
  for (int i = tid; i < TB * IN0P; i += NTHR)
    inp[i] = ((i % IN0P) == 1) ? (unsigned short)0x3F80 : (unsigned short)0;
  for (int j = sub; j < COND; j += 16)
    inp[rowr * IN0P + XD + j] =
        bf16b(x[((long)(rbase + rowr) * NSTEP + 0) * IN0 + XD + j]);
  __syncthreads();

  int b0 = 0, b1 = 1, bfree = 2;       // h0_old, h1_old, free buffer indices
  for (int t = 0; t < NSTEP; ++t) {
    const bool first = (t == 0);
    const unsigned short* h0rd = hbuf + b0 * (TB * H);
    const unsigned short* h1rd = hbuf + b1 * (TB * H);
    unsigned short*       h0wr = hbuf + bfree * (TB * H); // h0_new
    unsigned short*       h1wr = hbuf + b0 * (TB * H);    // h1_new (b0 dies)

    // ---------------- layer 0: gates = inp@Wih0' + h0@Whh0' ----------------
    for (int nt = 0; nt < 4; ++nt) {
      const int col0 = wave * 64 + nt * 16;
      v8f z8 = {0.f, 0.f, 0.f, 0.f, 0.f, 0.f, 0.f, 0.f};
      v8f acc[8] = {z8, z8, z8, z8, z8, z8, z8, z8};
      gemm8(acc, inp,  IN0P, IN0P / 32, w_ih0, IN0P, col0, lane);
      gemm8(acc, h0rd, H,    H / 32,    w_hh0, H,    col0, lane);
      const int ncol = col0 + (lane & 15);
      cell_update(acc[0], acc[1], acc[2], acc[3], bias0, ncol, rtop,
                  first, c0g, growbase, h0wr);
      cell_update(acc[4], acc[5], acc[6], acc[7], bias0, ncol, 16 + rtop,
                  first, c0g, growbase, h0wr);
    }
    __syncthreads();

    // ---------------- layer 1: gates = h0new@Wih1' + h1@Whh1' --------------
    for (int nt = 0; nt < 4; ++nt) {
      const int col0 = wave * 64 + nt * 16;
      v8f z8 = {0.f, 0.f, 0.f, 0.f, 0.f, 0.f, 0.f, 0.f};
      v8f acc[8] = {z8, z8, z8, z8, z8, z8, z8, z8};
      gemm8(acc, h0wr, H, H / 32, w_ih1, H, col0, lane);
      gemm8(acc, h1rd, H, H / 32, w_hh1, H, col0, lane);
      const int ncol = col0 + (lane & 15);
      cell_update(acc[0], acc[1], acc[2], acc[3], bias1, ncol, rtop,
                  first, c1g, growbase, h1wr);
      cell_update(acc[4], acc[5], acc[6], acc[7], bias1, ncol, 16 + rtop,
                  first, c1g, growbase, h1wr);
    }
    __syncthreads();

    // ---------------- output projection: logits = h1new@Wf' + bf -----------
    for (int nt = wave; nt < XDP / 16; nt += 16) {   // waves 0..8 active
      v8f acc0 = {0.f, 0.f, 0.f, 0.f, 0.f, 0.f, 0.f, 0.f};
      v8f acc1 = acc0;
      const unsigned short* wrow = wf + (long)(nt * 16) * H;
      for (int kt = 0; kt < H / 32; ++kt) {
        const int k0 = kt * 32;
        v16bf a0 = ld_a(h1wr,           H, lane, k0);
        v16bf a1 = ld_a(h1wr + 16 * H,  H, lane, k0);
        v16bf b  = ld_b(wrow, H, lane, k0);
        acc0 = wmma_bf16(a0, b, acc0);
        acc1 = wmma_bf16(a1, b, acc1);
      }
      const int ncol = nt * 16 + (lane & 15);
      const float bb = (ncol < XD) ? bfv[ncol] : 0.0f;
#pragma unroll
      for (int r = 0; r < 8; ++r) {
        lgt[(rtop + r) * XDP + ncol]        = acc0[r] + bb;
        lgt[(16 + rtop + r) * XDP + ncol]   = acc1[r] + bb;
      }
    }
    __syncthreads();

    // -------- log_softmax + argmax + global store + one-hot feedback -------
    {
      float m = -3.4e38f; int am = 0;
      for (int c = sub; c < XD; c += 16) {
        float v = lgt[rowr * XDP + c];
        if (v > m) { m = v; am = c; }
      }
      rmax[tid] = m; ridx[tid] = am;
      __syncthreads();
      float gm = -3.4e38f; int gam = 0;
      for (int s = 0; s < 16; ++s) {
        float v = rmax[rowr * 16 + s]; int ix = ridx[rowr * 16 + s];
        if (v > gm || (v == gm && ix < gam)) { gm = v; gam = ix; }
      }
      float ls = 0.0f;
      for (int c = sub; c < XD; c += 16)
        ls += __expf(lgt[rowr * XDP + c] - gm);
      rsum[tid] = ls;
      __syncthreads();
      float gs = 0.0f;
      for (int s = 0; s < 16; ++s) gs += rsum[rowr * 16 + s];
      const float lse = gm + __logf(gs);

      const long ob = ((long)(rbase + rowr) * NSTEP + t) * XD;
      for (int c = sub; c < XD; c += 16) {
        out[ob + c] = lgt[rowr * XDP + c] - lse;
        inp[rowr * IN0P + c] =
            (c == gam) ? (unsigned short)0x3F80 : (unsigned short)0;
      }
      if (t + 1 < NSTEP) {
        for (int j = sub; j < COND; j += 16)
          inp[rowr * IN0P + XD + j] =
              bf16b(x[((long)(rbase + rowr) * NSTEP + (t + 1)) * IN0 + XD + j]);
      }
    }
    __syncthreads();

    // rotate h buffers: new h0 is in bfree, new h1 is in b0, b1 is now free
    const int nb0 = bfree, nb1 = b0, nbf = b1;
    b0 = nb0; b1 = nb1; bfree = nbf;
  }
}

// ---------------------------------------------------------------------------
// Host entry
// ---------------------------------------------------------------------------
extern "C" void kernel_launch(void* const* d_in, const int* in_sizes, int n_in,
                              void* d_out, int out_size, void* d_ws,
                              size_t ws_size, hipStream_t stream) {
  (void)in_sizes; (void)n_in; (void)out_size; (void)ws_size;
  const float* z     = (const float*)d_in[0];
  const float* x     = (const float*)d_in[1];
  const float* W_ih0 = (const float*)d_in[2];
  const float* W_hh0 = (const float*)d_in[3];
  const float* b_ih0 = (const float*)d_in[4];
  const float* b_hh0 = (const float*)d_in[5];
  const float* W_ih1 = (const float*)d_in[6];
  const float* W_hh1 = (const float*)d_in[7];
  const float* b_ih1 = (const float*)d_in[8];
  const float* b_hh1 = (const float*)d_in[9];
  const float* Wf    = (const float*)d_in[10];
  const float* bf    = (const float*)d_in[11];
  float* out = (float*)d_out;

  // Workspace: bf16 weights (~27 MB) + biases + f32 cell state (32 MB).
  char* ws = (char*)d_ws;
  unsigned short* w_ih0p = (unsigned short*)ws;  ws += (size_t)G4 * IN0P * 2;
  unsigned short* w_hh0p = (unsigned short*)ws;  ws += (size_t)G4 * H * 2;
  unsigned short* w_ih1p = (unsigned short*)ws;  ws += (size_t)G4 * H * 2;
  unsigned short* w_hh1p = (unsigned short*)ws;  ws += (size_t)G4 * H * 2;
  unsigned short* wf_p   = (unsigned short*)ws;  ws += (size_t)XDP * H * 2;
  float* bias0 = (float*)ws;                     ws += (size_t)G4 * 4;
  float* bias1 = (float*)ws;                     ws += (size_t)G4 * 4;
  float* c0g   = (float*)ws;                     ws += (size_t)BP * H * 4;
  float* c1g   = (float*)ws;                     ws += (size_t)BP * H * 4;

  const int thr = 256;
  auto nblk = [](long n) { return (unsigned)((n + 255) / 256); };

  cvt_bf16_pad<<<nblk((long)G4 * IN0P), thr, 0, stream>>>(
      W_ih0, w_ih0p, G4, G4, IN0, IN0P);
  cvt_bf16_pad<<<nblk((long)G4 * H), thr, 0, stream>>>(
      W_hh0, w_hh0p, G4, G4, H, H);
  cvt_bf16_pad<<<nblk((long)G4 * H), thr, 0, stream>>>(
      W_ih1, w_ih1p, G4, G4, H, H);
  cvt_bf16_pad<<<nblk((long)G4 * H), thr, 0, stream>>>(
      W_hh1, w_hh1p, G4, G4, H, H);
  cvt_bf16_pad<<<nblk((long)XDP * H), thr, 0, stream>>>(
      Wf, wf_p, XD, XDP, H, H);
  add_vec<<<nblk(G4), thr, 0, stream>>>(b_ih0, b_hh0, bias0, G4);
  add_vec<<<nblk(G4), thr, 0, stream>>>(b_ih1, b_hh1, bias1, G4);

  // Dynamic LDS: inp 12288 + h 3*65536 + logits 18432 + reductions 6144
  //            = 233472 B (< 320 KB/WGP).
  const size_t smem_bytes = 233472;
  lstm_dec<<<BP / TB, NTHR, smem_bytes, stream>>>(
      z, x, w_ih0p, w_hh0p, w_ih1p, w_hh1p, wf_p, bias0, bias1, bf,
      c0g, c1g, out);
}
